// LSTMCell_88450556494554
// MI455X (gfx1250) — matrix operand order repaired and checked by reading, not compiled, not run on hardware
//
#include <hip/hip_runtime.h>
#include <hip/hip_bf16.h>

#define HID   1024
#define INP   512
#define BATCH 128
#define TIME  256
#define NBLK  512   // (HID/16) * (BATCH/16) persistent tile-waves

typedef __attribute__((ext_vector_type(16))) __bf16 v16bf;
typedef __attribute__((ext_vector_type(8)))  __bf16 v8bf;
typedef __attribute__((ext_vector_type(8)))  float  v8f;

union AFrag { v16bf v; v8bf h8[2]; };

// ---------------------------------------------------------------------------
// Phase 0 helpers
// ---------------------------------------------------------------------------
__global__ void lstm_cvt_bf16(const float* __restrict__ src,
                              __bf16* __restrict__ dst, int n) {
    int i = blockIdx.x * blockDim.x + threadIdx.x;
    if (i < n) dst[i] = (__bf16)src[i];
}

// h input is (H, B) fp32; store as (B, H) bf16 so A-fragment rows are batch rows.
__global__ void lstm_h_transpose(const float* __restrict__ h_in,
                                 __bf16* __restrict__ hbf) {
    int i = blockIdx.x * blockDim.x + threadIdx.x;   // i = n*B + m
    if (i < HID * BATCH) {
        int n = i >> 7;
        int m = i & (BATCH - 1);
        hbf[m * HID + n] = (__bf16)h_in[i];
    }
}

__global__ void lstm_bar_init(unsigned* bar) { if (threadIdx.x < 2) bar[threadIdx.x] = 0u; }

// ---------------------------------------------------------------------------
// Two-phase device-scope barrier (counter + generation), s_sleep backoff.
// One wave per block -> reconvergence after the `if` re-synchronizes lanes.
// ---------------------------------------------------------------------------
__device__ __forceinline__ void grid_sync(unsigned* __restrict__ cnt,
                                          unsigned* __restrict__ gen,
                                          unsigned target) {
    __threadfence();                      // release this step's h stores (agent scope)
    if (threadIdx.x == 0) {
        unsigned prev = __hip_atomic_fetch_add(cnt, 1u, __ATOMIC_ACQ_REL,
                                               __HIP_MEMORY_SCOPE_AGENT);
        if (prev == (unsigned)NBLK - 1u) {
            __hip_atomic_store(cnt, 0u, __ATOMIC_RELAXED, __HIP_MEMORY_SCOPE_AGENT);
            __hip_atomic_fetch_add(gen, 1u, __ATOMIC_RELEASE, __HIP_MEMORY_SCOPE_AGENT);
        } else {
            while (__hip_atomic_load(gen, __ATOMIC_ACQUIRE,
                                     __HIP_MEMORY_SCOPE_AGENT) < target)
                __builtin_amdgcn_s_sleep(1);
        }
    }
    __threadfence();                      // acquire side before next step's loads
}

// ---------------------------------------------------------------------------
// Persistent recurrence: one wave owns a 16(batch) x 16(hid) tile for all 256
// steps. grid = (HID/16, BATCH/16) = 512 single-wave blocks (no LDS -> all
// co-resident). Per step: z_q = h @ Wh_q^T + x_t @ Wx_q^T + b_q, K = 1024+512,
// 4 gate accumulators share one A fragment -> per-lane gate fusion, no shuffles.
// ---------------------------------------------------------------------------
__global__ __launch_bounds__(32)
void lstm_persistent(const __bf16* __restrict__ xbf,     // (B, T, I) bf16
                     __bf16* __restrict__ hb0,           // (B, H) bf16 ping
                     __bf16* __restrict__ hb1,           // (B, H) bf16 pong
                     float* __restrict__ cws,            // (H, B) fp32, in-place
                     const __bf16* __restrict__ whbf,    // 4 x (H, H) bf16
                     const __bf16* __restrict__ wxbf,    // 4 x (H, I) bf16
                     const float* __restrict__ bg, const float* __restrict__ bi,
                     const float* __restrict__ bfv, const float* __restrict__ bo,
                     float* __restrict__ out,            // (H, B) fp32
                     unsigned* __restrict__ bar) {
    const int lane = threadIdx.x;
    const int half = lane >> 4;
    const int ln   = lane & 15;
    const int h0   = blockIdx.x * 16;  // hid tile base
    const int m0   = blockIdx.y * 16;  // batch tile base
    const int n    = h0 + ln;          // this lane's hid index (C/D: N = lane%16)

    // t-invariant state, hoisted
    const float vbg = bg[n], vbi = bi[n], vbf = bfv[n], vbo = bo[n];
    float* crow = cws + (size_t)n * BATCH + m0 + 8 * half;  // 8 contiguous floats
    const __bf16* bh0 = whbf + 0 * (size_t)(HID * HID) + (size_t)n * HID;
    const __bf16* bh1 = whbf + 1 * (size_t)(HID * HID) + (size_t)n * HID;
    const __bf16* bh2 = whbf + 2 * (size_t)(HID * HID) + (size_t)n * HID;
    const __bf16* bh3 = whbf + 3 * (size_t)(HID * HID) + (size_t)n * HID;
    const __bf16* bx0 = wxbf + 0 * (size_t)(HID * INP) + (size_t)n * INP;
    const __bf16* bx1 = wxbf + 1 * (size_t)(HID * INP) + (size_t)n * INP;
    const __bf16* bx2 = wxbf + 2 * (size_t)(HID * INP) + (size_t)n * INP;
    const __bf16* bx3 = wxbf + 3 * (size_t)(HID * INP) + (size_t)n * INP;
    const __bf16* rowXbase = xbf + (size_t)(m0 + ln) * TIME * INP;
    unsigned* bcnt = bar;
    unsigned* bgen = bar + 1;

    for (int t = 0; t < TIME; ++t) {
        const __bf16* hin  = (t & 1) ? hb1 : hb0;
        __bf16*       hout = (t & 1) ? hb0 : hb1;
        const __bf16* rowA = hin + (size_t)(m0 + ln) * HID;
        const __bf16* rowX = rowXbase + (size_t)t * INP;

        v8f acc0 = {}, acc1 = {}, acc2 = {}, acc3 = {};

        // ---- K = 0..1023 : recurrent contraction (A: 16x32 bf16, contiguous k = 16*half + s, same as B) ----
#pragma unroll 4
        for (int k = 0; k < HID; k += 32) {
            AFrag a;
            a.h8[0] = *(const v8bf*)(rowA + k + 16 * half);
            a.h8[1] = *(const v8bf*)(rowA + k + 16 * half + 8);
            __builtin_prefetch(bh0 + k + 64, 0, 1);
            v16bf b0 = *(const v16bf*)(bh0 + k + 16 * half);
            v16bf b1 = *(const v16bf*)(bh1 + k + 16 * half);
            v16bf b2 = *(const v16bf*)(bh2 + k + 16 * half);
            v16bf b3 = *(const v16bf*)(bh3 + k + 16 * half);
            acc0 = __builtin_amdgcn_wmma_f32_16x16x32_bf16(false, a.v, false, b0, (short)0, acc0, false, false);
            acc1 = __builtin_amdgcn_wmma_f32_16x16x32_bf16(false, a.v, false, b1, (short)0, acc1, false, false);
            acc2 = __builtin_amdgcn_wmma_f32_16x16x32_bf16(false, a.v, false, b2, (short)0, acc2, false, false);
            acc3 = __builtin_amdgcn_wmma_f32_16x16x32_bf16(false, a.v, false, b3, (short)0, acc3, false, false);
        }

        // ---- K = 0..511 : fused input projection for timestep t ----
#pragma unroll 4
        for (int k = 0; k < INP; k += 32) {
            AFrag a;
            a.h8[0] = *(const v8bf*)(rowX + k + 16 * half);
            a.h8[1] = *(const v8bf*)(rowX + k + 16 * half + 8);
            __builtin_prefetch(bx0 + k + 64, 0, 1);
            v16bf b0 = *(const v16bf*)(bx0 + k + 16 * half);
            v16bf b1 = *(const v16bf*)(bx1 + k + 16 * half);
            v16bf b2 = *(const v16bf*)(bx2 + k + 16 * half);
            v16bf b3 = *(const v16bf*)(bx3 + k + 16 * half);
            acc0 = __builtin_amdgcn_wmma_f32_16x16x32_bf16(false, a.v, false, b0, (short)0, acc0, false, false);
            acc1 = __builtin_amdgcn_wmma_f32_16x16x32_bf16(false, a.v, false, b1, (short)0, acc1, false, false);
            acc2 = __builtin_amdgcn_wmma_f32_16x16x32_bf16(false, a.v, false, b2, (short)0, acc2, false, false);
            acc3 = __builtin_amdgcn_wmma_f32_16x16x32_bf16(false, a.v, false, b3, (short)0, acc3, false, false);
        }

        // ---- Gates + cell update; C/D element = (M = r + 8*half, N = ln) ----
#pragma unroll
        for (int r = 0; r < 8; ++r) {
            float zg = acc0[r] + vbg;
            float zi = acc1[r] + vbi;
            float zf = acc2[r] + vbf;
            float zo = acc3[r] + vbo;
            float gv = tanhf(zg);
            float iv = 1.0f / (1.0f + __expf(-zi));
            float fv = 1.0f / (1.0f + __expf(-zf));
            float ov = 1.0f / (1.0f + __expf(-zo));
            float cn = gv * iv + crow[r] * fv;
            crow[r] = cn;
            float hn = tanhf(cn) * ov;
            hout[(size_t)(m0 + 8 * half + r) * HID + n] = (__bf16)hn;
            if (t == TIME - 1)
                out[(size_t)n * BATCH + m0 + 8 * half + r] = hn;  // fp32 (H,B) output
        }

        if (t + 1 < TIME) grid_sync(bcnt, bgen, (unsigned)(t + 1));
    }
}

// ---------------------------------------------------------------------------
extern "C" void kernel_launch(void* const* d_in, const int* in_sizes, int n_in,
                              void* d_out, int out_size, void* d_ws, size_t ws_size,
                              hipStream_t stream) {
    (void)in_sizes; (void)n_in; (void)out_size; (void)ws_size;

    const float* x  = (const float*)d_in[0];
    const float* c0 = (const float*)d_in[1];
    const float* h0 = (const float*)d_in[2];
    const float* Wx[4] = { (const float*)d_in[3], (const float*)d_in[4],
                           (const float*)d_in[5], (const float*)d_in[6] };   // g,i,f,o
    const float* Wh[4] = { (const float*)d_in[7], (const float*)d_in[8],
                           (const float*)d_in[9], (const float*)d_in[10] };  // g,i,f,o
    const float* bg = (const float*)d_in[11];
    const float* bi = (const float*)d_in[12];
    const float* bf = (const float*)d_in[13];
    const float* bo = (const float*)d_in[14];

    // Workspace layout (256B-aligned), ~45.5 MB
    char* ws = (char*)d_ws;
    __bf16*   whbf = (__bf16*)(ws);                               //  8 MB: 4 x (H,H)
    __bf16*   wxbf = (__bf16*)(ws + ((size_t)8  << 20));          //  4 MB: 4 x (H,I)
    __bf16*   xbf  = (__bf16*)(ws + ((size_t)12 << 20));          // 32 MB: (B,T,I)
    __bf16*   hbf0 = (__bf16*)(ws + ((size_t)44 << 20));          // 256 KB
    __bf16*   hbf1 = (__bf16*)(ws + ((size_t)44 << 20) + 262144); // 256 KB
    float*    cws  = (float*) (ws + ((size_t)45 << 20));          // 512 KB
    unsigned* bar  = (unsigned*)(ws + ((size_t)45 << 20) + 524288); // {cnt, gen}

    // Phase 0: bf16 conversion + state seeding (d_ws is poisoned -> init everything)
    const int nWh = HID * HID;   // 1M
    const int nWx = HID * INP;   // 512K
    for (int q = 0; q < 4; ++q) {
        lstm_cvt_bf16<<<nWh / 256, 256, 0, stream>>>(Wh[q], whbf + (size_t)q * nWh, nWh);
        lstm_cvt_bf16<<<nWx / 256, 256, 0, stream>>>(Wx[q], wxbf + (size_t)q * nWx, nWx);
    }
    const int nX = BATCH * TIME * INP;  // 16M
    lstm_cvt_bf16<<<nX / 256, 256, 0, stream>>>(x, xbf, nX);
    lstm_h_transpose<<<(HID * BATCH) / 256, 256, 0, stream>>>(h0, hbf0);
    hipMemcpyAsync(cws, c0, (size_t)HID * BATCH * sizeof(float),
                   hipMemcpyDeviceToDevice, stream);
    lstm_bar_init<<<1, 32, 0, stream>>>(bar);

    // Phase 1: single persistent kernel runs the full T=256 recurrence with
    // device-scope barriers between steps (512 single-wave blocks, all resident).
    dim3 grid(HID / 16, BATCH / 16);   // 64 x 8 = NBLK
    lstm_persistent<<<grid, 32, 0, stream>>>(xbf, hbf0, hbf1, cws, whbf, wxbf,
                                             bg, bi, bf, bo, (float*)d_out, bar);
}
